// Cora_GCN_Method_70755291234414
// MI455X (gfx1250) — matrix operand order repaired and checked by reading, not compiled, not run on hardware
//
#include <hip/hip_runtime.h>
#include <hip/hip_fp16.h>

typedef __attribute__((ext_vector_type(16))) _Float16 v16h;
typedef __attribute__((ext_vector_type(8)))  float    v8f;

#define NFEAT 512
#define H1    300
#define H2    200
#define NCLS  40
#define H1P   320   // padded: %64==0 (N), %32==0 (K)
#define H2P   256
#define NCP   64

// ---------------- elementwise f32 -> f16 ----------------
__global__ __launch_bounds__(256) void f32_to_f16_kernel(
    const float* __restrict__ in, _Float16* __restrict__ out, size_t n) {
  size_t i = (size_t)blockIdx.x * blockDim.x + threadIdx.x;
  if (i < n) out[i] = (_Float16)in[i];
}

// ---------------- transpose + pad weights: wt[n][k] = w[k][n] ----------------
__global__ __launch_bounds__(256) void transpose_pad_f16_kernel(
    const float* __restrict__ w, _Float16* __restrict__ wt,
    int K, int N, int Kp, int Np) {
  size_t i = (size_t)blockIdx.x * blockDim.x + threadIdx.x;
  if (i >= (size_t)Kp * Np) return;
  int k = (int)(i % Kp);
  int n = (int)(i / Kp);
  _Float16 v = (_Float16)0.0f;
  if (k < K && n < N) v = (_Float16)w[(size_t)k * N + n];
  wt[(size_t)n * Kp + k] = v;
}

// ---------------- WMMA GEMM: C[M][Np] = A[M][K] * Bt[Np][K]^T ----------------
// K % 32 == 0, Np % 64 == 0, M % 16 == 0.
// Block: 256 threads = 8 waves. Block computes 128 rows x 64 cols.
// B strip (64 x K) staged in LDS once per block (stride padded +8 halves =>
// conflict-free b128 reads). A double-buffered a full k-step ahead; all 4 B
// fragments batched into registers before the 4 back-to-back WMMAs.
__global__ __launch_bounds__(256) void gemm_wmma_f16(
    const _Float16* __restrict__ A,   // [M][K] row-major (f16)
    const _Float16* __restrict__ Bt,  // [Np][K] row-major = B transposed (f16)
    float* __restrict__ C,            // [M][Np] (f32)
    int M, int K, int Np) {
  extern __shared__ _Float16 Bs[];    // 64 * (K+8) halves
  const int Ks     = K + 8;           // padded LDS row stride (halves)
  const int lane   = threadIdx.x & 31;
  const int wave   = threadIdx.x >> 5;
  const int tileM  = blockIdx.x * 8 + wave;
  const int half16 = lane >> 4;       // 0: lanes 0-15, 1: lanes 16-31
  const int lm     = lane & 15;
  const int n0     = blockIdx.y * 64;

  // ---- cooperative global->LDS stage of the 64-row B strip ----
  {
    const int chunksPerRow = K >> 3;              // uint4 chunks per row
    const int totalChunks  = 64 * chunksPerRow;
    for (int idx = threadIdx.x; idx < totalChunks; idx += 256) {
      int n  = idx / chunksPerRow;
      int kc = (idx - n * chunksPerRow) << 3;
      *(uint4*)(Bs + n * Ks + kc) =
          *(const uint4*)(Bt + (size_t)(n0 + n) * K + kc);
    }
  }
  __syncthreads();

  if (tileM * 16 < M) {               // wave-uniform: EXEC stays all-1s for WMMA
    const _Float16* arow = A + (size_t)(tileM * 16 + lm) * K;

    union frag { v16h v; uint4 q[2]; };
    v8f acc[4] = {v8f{}, v8f{}, v8f{}, v8f{}};

    // prime A fragment for kk = 0
    frag a_cur;
    {
      const _Float16* ap = arow + half16 * 8;
      a_cur.q[0] = *(const uint4*)(ap);
      a_cur.q[1] = *(const uint4*)(ap + 16);
    }

    for (int kk = 0; kk < K; kk += 32) {
      // prefetch next A fragment (dummy re-load of kk=0 on last iter)
      frag a_nxt;
      {
        int kn = (kk + 32 < K) ? (kk + 32) : 0;
        const _Float16* ap = arow + kn + half16 * 8;
        a_nxt.q[0] = *(const uint4*)(ap);
        a_nxt.q[1] = *(const uint4*)(ap + 16);
      }
      // batch all 4 B fragments from LDS, then issue 4 independent WMMAs
      frag b[4];
#pragma unroll
      for (int t = 0; t < 4; ++t) {
        const _Float16* bp = Bs + (t * 16 + lm) * Ks + kk + half16 * 8;
        b[t].q[0] = *(const uint4*)(bp);
        b[t].q[1] = *(const uint4*)(bp + 16);
      }
#pragma unroll
      for (int t = 0; t < 4; ++t) {
        acc[t] = __builtin_amdgcn_wmma_f32_16x16x32_f16(
            false, a_cur.v, false, b[t].v, (short)0, acc[t], false, false);
      }
      a_cur = a_nxt;
    }

    // C/D layout: lane lm = col; VGPR e -> row = e + half16*8
#pragma unroll
    for (int t = 0; t < 4; ++t) {
      int n = n0 + t * 16 + lm;
#pragma unroll
      for (int e = 0; e < 8; ++e) {
        int m = tileM * 16 + half16 * 8 + e;
        C[(size_t)m * Np + n] = acc[t][e];
      }
    }
  }
}

// ---------------- edge scatter: G[dst] += S[src] * w  (float4 chunks) ----------------
__global__ __launch_bounds__(256) void scatter_kernel(
    const float* __restrict__ S, const int* __restrict__ src,
    const int* __restrict__ dst, const float* __restrict__ ew,
    float* __restrict__ G, unsigned E, unsigned F4, int stride) {
  unsigned tid = blockIdx.x * blockDim.x + threadIdx.x;
  if (tid >= E * F4) return;
  unsigned e = tid / F4;              // consecutive lanes share an edge -> coalesced row reads
  int c = (int)(tid - e * F4) * 4;
  int s = src[e];
  int d = dst[e];
  float w = ew[e];
  const float4 v = *(const float4*)(S + (size_t)s * stride + c);
  float* g = G + (size_t)d * stride + c;
  atomicAdd(g + 0, v.x * w);
  atomicAdd(g + 1, v.y * w);
  atomicAdd(g + 2, v.z * w);
  atomicAdd(g + 3, v.w * w);
}

// ---------------- bias + relu + convert to f16 (zero the pad columns) ----------------
__global__ __launch_bounds__(256) void bias_relu_f16_kernel(
    const float* __restrict__ G, const float* __restrict__ bias,
    _Float16* __restrict__ H, int M, int out, int stride) {
  size_t i = (size_t)blockIdx.x * blockDim.x + threadIdx.x;
  if (i >= (size_t)M * stride) return;
  int j = (int)(i % stride);
  float v = 0.0f;
  if (j < out) {
    v = G[i] + bias[j];
    v = v > 0.0f ? v : 0.0f;
  }
  H[i] = (_Float16)v;
}

// ---------------- per-node bias + softmax over 40 classes ----------------
__global__ __launch_bounds__(256) void softmax_kernel(
    const float* __restrict__ G, const float* __restrict__ bias,
    float* __restrict__ out, int M) {
  int i = blockIdx.x * blockDim.x + threadIdx.x;
  if (i >= M) return;
  const float* g = G + (size_t)i * NCP;
  float l[NCLS];
  float mx = -1e30f;
#pragma unroll
  for (int j = 0; j < NCLS; ++j) {
    float v = g[j] + bias[j];
    l[j] = v;
    mx = fmaxf(mx, v);
  }
  float s = 0.0f;
#pragma unroll
  for (int j = 0; j < NCLS; ++j) {
    float e = __expf(l[j] - mx);
    l[j] = e;
    s += e;
  }
  float inv = 1.0f / s;
  float* o = out + (size_t)i * NCLS;
#pragma unroll
  for (int j = 0; j < NCLS; ++j) o[j] = l[j] * inv;
}

extern "C" void kernel_launch(void* const* d_in, const int* in_sizes, int n_in,
                              void* d_out, int out_size, void* d_ws, size_t ws_size,
                              hipStream_t stream) {
  const float* x  = (const float*)d_in[0];
  const int*   ei = (const int*)d_in[1];
  const float* ew = (const float*)d_in[2];
  const float* w1 = (const float*)d_in[3];
  const float* b1 = (const float*)d_in[4];
  const float* w2 = (const float*)d_in[5];
  const float* b2 = (const float*)d_in[6];
  const float* w3 = (const float*)d_in[7];
  const float* b3 = (const float*)d_in[8];
  float* out = (float*)d_out;

  const unsigned E = (unsigned)in_sizes[2];
  const int M = in_sizes[0] / NFEAT;      // 100000, multiple of 16
  const int* src = ei;
  const int* dst = ei + E;

  // -------- workspace carve-out (aliased across layers) --------
  char* ws = (char*)d_ws;
  size_t off = 0;
  auto alloc = [&](size_t bytes) {
    char* p = ws + off;
    off += (bytes + 255) & ~(size_t)255;
    return p;
  };
  _Float16* xf16 = (_Float16*)alloc((size_t)M * NFEAT * 2);     // 102.4 MB
  _Float16* w1t  = (_Float16*)alloc((size_t)H1P * NFEAT * 2);
  _Float16* w2t  = (_Float16*)alloc((size_t)H2P * H1P * 2);
  _Float16* w3t  = (_Float16*)alloc((size_t)NCP * H2P * 2);
  float*    S    = (float*)alloc((size_t)M * H1P * 4);          // supports (128 MB, reused)
  float*    G    = (float*)alloc((size_t)M * H1P * 4);          // aggregates (128 MB, reused)
  _Float16* Hh   = (_Float16*)alloc((size_t)M * H1P * 2);       // f16 activations (reused)

  const int mtiles8 = (M / 16 + 7) / 8;
  auto ldsBytes = [](int K) { return (size_t)64 * (K + 8) * sizeof(_Float16); };

  // -------- input / weight conversion --------
  { size_t n = (size_t)M * NFEAT;
    f32_to_f16_kernel<<<(n + 255) / 256, 256, 0, stream>>>(x, xf16, n); }
  { size_t n = (size_t)H1P * NFEAT;
    transpose_pad_f16_kernel<<<(n + 255) / 256, 256, 0, stream>>>(w1, w1t, NFEAT, H1, NFEAT, H1P); }
  { size_t n = (size_t)H2P * H1P;
    transpose_pad_f16_kernel<<<(n + 255) / 256, 256, 0, stream>>>(w2, w2t, H1, H2, H1P, H2P); }
  { size_t n = (size_t)NCP * H2P;
    transpose_pad_f16_kernel<<<(n + 255) / 256, 256, 0, stream>>>(w3, w3t, H2, NCLS, H2P, NCP); }

  // -------- layer 1 --------
  gemm_wmma_f16<<<dim3(mtiles8, H1P / 64), 256, ldsBytes(NFEAT), stream>>>(
      xf16, w1t, S, M, NFEAT, H1P);
  hipMemsetAsync(G, 0, (size_t)M * H1P * 4, stream);
  { size_t n = (size_t)E * (H1 / 4);
    scatter_kernel<<<(n + 255) / 256, 256, 0, stream>>>(S, src, dst, ew, G, E, H1 / 4, H1P); }
  { size_t n = (size_t)M * H1P;
    bias_relu_f16_kernel<<<(n + 255) / 256, 256, 0, stream>>>(G, b1, Hh, M, H1, H1P); }

  // -------- layer 2 --------
  gemm_wmma_f16<<<dim3(mtiles8, H2P / 64), 256, ldsBytes(H1P), stream>>>(
      Hh, w2t, S, M, H1P, H2P);
  hipMemsetAsync(G, 0, (size_t)M * H2P * 4, stream);
  { size_t n = (size_t)E * (H2 / 4);
    scatter_kernel<<<(n + 255) / 256, 256, 0, stream>>>(S, src, dst, ew, G, E, H2 / 4, H2P); }
  { size_t n = (size_t)M * H2P;
    bias_relu_f16_kernel<<<(n + 255) / 256, 256, 0, stream>>>(G, b2, Hh, M, H2, H2P); }

  // -------- layer 3 + softmax --------
  gemm_wmma_f16<<<dim3(mtiles8, NCP / 64), 256, ldsBytes(H2P), stream>>>(
      Hh, w3t, S, M, H2P, NCP);
  hipMemsetAsync(G, 0, (size_t)M * NCP * 4, stream);
  { size_t n = (size_t)E * (NCLS / 4);
    scatter_kernel<<<(n + 255) / 256, 256, 0, stream>>>(S, src, dst, ew, G, E, NCLS / 4, NCP); }
  softmax_kernel<<<(M + 255) / 256, 256, 0, stream>>>(G, b3, out, M);
}
